// CustomCrossAttention_76845554860870
// MI455X (gfx1250) — compile-verified, matrix operand order checked by reading, not run on hardware
//
#include <hip/hip_runtime.h>
#include <hip/hip_bf16.h>
#include <math.h>

// ---------------------------------------------------------------------------
// CustomCrossAttention for MI455X (gfx1250): bf16 WMMA pipeline with
// async global->LDS staging (ASYNCcnt) and fused epilogues.
// b=16, n=4096, j=77, d_model=1024, heads=16, dim_head=64.
// Memory-bound (~1GB traffic, ~440 GFLOP) -> bf16 intermediates, WMMA GEMMs,
// global_load_async_to_lds_b128 double-buffered staging.
// ---------------------------------------------------------------------------

typedef __attribute__((ext_vector_type(16))) __bf16 v16bf;
typedef __attribute__((ext_vector_type(8)))  float  v8f;

constexpr int BB = 16;
constexpr int NN = 4096;
constexpr int JJ = 77;
constexpr int JP = 96;          // j padded to multiple of 32 (WMMA K-dim)
constexpr int JT = JP / 16;     // 6 j-tiles
constexpr int QD = 1024;
constexpr int HH = 16;
constexpr int DH = 64;
constexpr float ATTN_SCALE = 0.125f;  // 64^-0.5

__device__ __forceinline__ float gelu_exact(float v) {
  return 0.5f * v * (1.0f + erff(v * 0.70710678118654752f));
}

__device__ __forceinline__ v8f wmma_bf16(v16bf a, v16bf b, v8f c) {
  // (neg_a, A, neg_b, B, c_mod, C, reuse_a, reuse_b)
  return __builtin_amdgcn_wmma_f32_16x16x32_bf16(false, a, false, b, (short)0, c,
                                                 false, false);
}

// Fragment loader for 16x32 bf16 A (or B presented as [n][k]) tiles from a
// row-major [rows][ld] matrix with the K dimension contiguous.
// ISA 16-bit A-layout: lanes 0-15 row M=lane hold K = {0..7, 16..23};
// lanes 16-31 same row hold K = {8..15, 24..31}  (two contiguous 8-elt runs).
__device__ __forceinline__ v16bf frag_ld(const __bf16* p, int ld, int r0, int k0) {
  int lane = threadIdx.x & 31;
  int idx  = lane & 15;
  int kb   = k0 + ((lane >> 4) << 3);
  const __bf16* q = p + (size_t)(r0 + idx) * ld + kb;
  v16bf f;
#pragma unroll
  for (int e = 0; e < 8; ++e) { f[e] = q[e]; f[e + 8] = q[16 + e]; }
  return f;
}

__device__ __forceinline__ v16bf frag_ld_guard(const __bf16* p, int ld, int r0,
                                               int k0, int rmax) {
  int lane = threadIdx.x & 31;
  int idx  = lane & 15;
  int r    = r0 + idx;
  int kb   = k0 + ((lane >> 4) << 3);
  v16bf f;
  if (r < rmax) {
    const __bf16* q = p + (size_t)r * ld + kb;
#pragma unroll
    for (int e = 0; e < 8; ++e) { f[e] = q[e]; f[e + 8] = q[16 + e]; }
  } else {
#pragma unroll
    for (int e = 0; e < 16; ++e) f[e] = (__bf16)0.0f;
  }
  return f;
}

// ---------------------------------------------------------------------------
// Async global->LDS 16-byte copy (CDNA5 ASYNCcnt path, GVS addressing):
//   copies 16 bytes from global saddr+vaddr to LDS address vdst   (8 bf16 per op)
// ---------------------------------------------------------------------------
__device__ __forceinline__ void async_ld_b128(void* lds_dst, const void* gbase,
                                              uint32_t gbyteoff) {
  uint32_t lds = (uint32_t)(uintptr_t)lds_dst;  // low 32 bits = LDS offset
  asm volatile("global_load_async_to_lds_b128 %0, %1, %2"
               :: "v"(lds), "v"(gbyteoff), "s"(gbase)
               : "memory");
}

__device__ __forceinline__ void wait_async0() {
  asm volatile("s_wait_asynccnt 0x0" ::: "memory");
}

// Stage a 128-row x 32-col bf16 tile (k contiguous) into LDS asynchronously.
// 512 chunks of 16B; 256 threads issue 2 async ops each.
__device__ __forceinline__ void async_tile(const __bf16* g, int ld, int r0, int kb,
                                           __bf16* lbuf, int t) {
#pragma unroll
  for (int c = t; c < 512; c += 256) {
    int row = c >> 2, off = (c & 3) << 3;
    uint32_t go = (uint32_t)(((uint32_t)(r0 + row) * (uint32_t)ld +
                              (uint32_t)(kb + off)) * 2u);
    async_ld_b128(lbuf + row * 32 + off, g, go);
  }
}

// ---------------------------------------------------------------------------
// Weight convert + transpose: w[k][n] (f32) -> wt[n][k] (bf16, k contiguous)
// ---------------------------------------------------------------------------
__global__ void cvt_transpose(const float* __restrict__ w, __bf16* __restrict__ wt,
                              int K, int Nn) {
  size_t i = (size_t)blockIdx.x * blockDim.x + threadIdx.x;
  size_t total = (size_t)K * Nn;
  if (i >= total) return;
  int n = (int)(i / K);
  int k = (int)(i % K);
  wt[i] = (__bf16)w[(size_t)k * Nn + n];
}

// ---------------------------------------------------------------------------
// Context/type-embedding MLP (1232 rows): te lookup, gelu MLP, gating.
// ---------------------------------------------------------------------------
__global__ void prep_context(const float* __restrict__ ctx, const int* __restrict__ ctype,
                             const float* __restrict__ tt_emb,
                             const float* __restrict__ m1w, const float* __restrict__ m1b,
                             const float* __restrict__ m2w, const float* __restrict__ m2b,
                             const float* __restrict__ gA, const float* __restrict__ gB,
                             __bf16* __restrict__ ctx_bf, __bf16* __restrict__ kin_bf) {
  __shared__ float cpt[QD];
  __shared__ float te_s[QD];
  __shared__ float hid[QD / 2];
  int row = blockIdx.x;                 // b*JJ + j
  int t = threadIdx.x;
  int ct = ctype[row]; if (ct < 0) ct = 0;
  const float* crow = ctx + (size_t)row * QD;
  const float* terow = tt_emb + (size_t)ct * QD;
  for (int i = t; i < QD; i += 256) {
    float c = crow[i], e = terow[i];
    te_s[i] = e;
    cpt[i]  = c + e;
    ctx_bf[(size_t)row * QD + i] = (__bf16)c;
  }
  __syncthreads();
  for (int o = t; o < QD / 2; o += 256) {
    float acc = m1b[o];
    for (int i = 0; i < QD; ++i) acc += cpt[i] * m1w[(size_t)i * (QD / 2) + o];
    hid[o] = gelu_exact(acc);
  }
  __syncthreads();
  for (int o = t; o < QD; o += 256) {
    float acc = m2b[o];
    for (int i = 0; i < QD / 2; ++i) acc += hid[i] * m2w[(size_t)i * QD + o];
    float kin = crow[o] + te_s[o] * gA[o] + acc * gB[o];
    kin_bf[(size_t)row * QD + o] = (__bf16)kin;
  }
}

// ---------------------------------------------------------------------------
// Progress embedding (16 rows): pe = relu(p*p1w + p1b) @ p2w + p2b
// ---------------------------------------------------------------------------
__global__ void prep_pe(const float* __restrict__ progress,
                        const float* __restrict__ p1w, const float* __restrict__ p1b,
                        const float* __restrict__ p2w, const float* __restrict__ p2b,
                        float* __restrict__ pe) {
  __shared__ float hid[QD / 2];
  int b = blockIdx.x;
  float p = progress[b];
  int t = threadIdx.x;
  for (int o = t; o < QD / 2; o += 256) {
    float v = p * p1w[o] + p1b[o];
    hid[o] = v > 0.f ? v : 0.f;
  }
  __syncthreads();
  for (int o = t; o < QD; o += 256) {
    float acc = p2b[o];
    for (int i = 0; i < QD / 2; ++i) acc += hid[i] * p2w[(size_t)i * QD + o];
    pe[(size_t)b * QD + o] = acc;
  }
}

// ---------------------------------------------------------------------------
// xpe = bf16(x + pe[b])  (input to pe MLP layer 1)
// ---------------------------------------------------------------------------
__global__ void make_xpe(const float* __restrict__ x, const float* __restrict__ pe,
                         __bf16* __restrict__ xpe) {
  size_t i = (size_t)blockIdx.x * blockDim.x + threadIdx.x;
  int d = (int)(i % QD);
  int b = (int)(i / ((size_t)NN * QD));
  xpe[i] = (__bf16)(x[i] + pe[(size_t)b * QD + d]);
}

// ---------------------------------------------------------------------------
// K/V projections, one wave per 16(j) x 64(d) tile per head.
//   Kt  [b][h][jp][d]  (j rows, d contiguous)   for QK^T B-fragments
//   Vst [b][h][d][jp]  (d rows, j contiguous)   for P@V  B-fragments
// Rows j>=77 zero-filled via guarded A fragments.
// ---------------------------------------------------------------------------
__global__ void kv_gemm(const __bf16* __restrict__ kin, const __bf16* __restrict__ ctxb,
                        const __bf16* __restrict__ wkt, const __bf16* __restrict__ wvt,
                        __bf16* __restrict__ Kt, __bf16* __restrict__ Vst) {
  int b = blockIdx.x / JT, jt = blockIdx.x % JT;
  int h = blockIdx.y;
  int r0 = b * JJ + jt * 16;
  int rmax = b * JJ + JJ;
  v8f accK[4], accV[4];
#pragma unroll
  for (int s = 0; s < 4; ++s)
#pragma unroll
    for (int r = 0; r < 8; ++r) { accK[s][r] = 0.f; accV[s][r] = 0.f; }

  for (int kb = 0; kb < QD; kb += 32) {
    v16bf aK = frag_ld_guard(kin,  QD, r0, kb, rmax);
    v16bf aV = frag_ld_guard(ctxb, QD, r0, kb, rmax);
#pragma unroll
    for (int s = 0; s < 4; ++s) {
      v16bf bK = frag_ld(wkt, QD, h * DH + s * 16, kb);
      v16bf bV = frag_ld(wvt, QD, h * DH + s * 16, kb);
      accK[s] = wmma_bf16(aK, bK, accK[s]);
      accV[s] = wmma_bf16(aV, bV, accV[s]);
    }
  }
  int lane = threadIdx.x & 31;
  int idx = lane & 15, half = lane >> 4;
  size_t bh = (size_t)(b * HH + h);
#pragma unroll
  for (int s = 0; s < 4; ++s)
#pragma unroll
    for (int r = 0; r < 8; ++r) {
      int j = jt * 16 + r + 8 * half;
      int d = s * 16 + idx;
      Kt[(bh * JP + j) * DH + d]  = (__bf16)accK[s][r];
      Vst[(bh * DH + d) * JP + j] = (__bf16)accV[s][r];
    }
}

// ---------------------------------------------------------------------------
// 128x128-tile bf16 GEMM: C = A(MxK) @ Bt(NxK)^T.
// Double-buffered async global->LDS staging; 8 waves, each 16 rows x 128 cols
// (8 WMMA accumulators); fused epilogues:
//   mode 0: bf16( gelu(acc + bias[n]) )                    -> Cbf
//   mode 1: bf16( x + pe[b]*gA + (acc+bias)*gB )           -> Cbf (q input)
//   mode 2: Q head reshape: Cbf[b][h][i][d] = bf16(acc)
//   mode 3: f32( acc + bias[n] )                           -> Cf  (final out)
// ---------------------------------------------------------------------------
__global__ __launch_bounds__(256)
void gemm_bf16(const __bf16* __restrict__ A, const __bf16* __restrict__ Bt,
               int Kd, int Nn, int mode,
               const float* __restrict__ bias,
               const float* __restrict__ xf, const float* __restrict__ pe,
               const float* __restrict__ gA, const float* __restrict__ gB,
               __bf16* __restrict__ Cbf, float* __restrict__ Cf) {
  __shared__ __align__(16) __bf16 As[2][128 * 32];
  __shared__ __align__(16) __bf16 Bs[2][128 * 32];
  int m0 = blockIdx.x * 128, n0 = blockIdx.y * 128;
  int t = threadIdx.x, w = t >> 5;
  v8f acc[8];
#pragma unroll
  for (int s = 0; s < 8; ++s)
#pragma unroll
    for (int r = 0; r < 8; ++r) acc[s][r] = 0.f;

  // prologue: stage tile 0
  async_tile(A, Kd, m0, 0, As[0], t);
  async_tile(Bt, Kd, n0, 0, Bs[0], t);

  int nk = Kd / 32;
  for (int i = 0; i < nk; ++i) {
    int cur = i & 1;
    wait_async0();        // own tile-i chunks landed
    __syncthreads();      // everyone's tile-i staged; prev compute done
    if (i + 1 < nk) {     // overlap: stream tile i+1 while computing tile i
      async_tile(A, Kd, m0, (i + 1) * 32, As[cur ^ 1], t);
      async_tile(Bt, Kd, n0, (i + 1) * 32, Bs[cur ^ 1], t);
    }
    v16bf a = frag_ld(As[cur], 32, w << 4, 0);
#pragma unroll
    for (int s = 0; s < 8; ++s) {
      v16bf bb = frag_ld(Bs[cur], 32, s * 16, 0);
      acc[s] = wmma_bf16(a, bb, acc[s]);
    }
    __syncthreads();      // done reading buf[cur] before it is refilled
  }

  int lane = t & 31;
  int idx = lane & 15, half = lane >> 4;
#pragma unroll
  for (int s = 0; s < 8; ++s)
#pragma unroll
    for (int r = 0; r < 8; ++r) {
      size_t m = (size_t)m0 + (w << 4) + r + 8 * half;
      int n = n0 + s * 16 + idx;
      float v = acc[s][r];
      if (mode == 0) {
        Cbf[m * Nn + n] = (__bf16)gelu_exact(v + bias[n]);
      } else if (mode == 1) {
        v += bias[n];
        int b = (int)(m / NN);
        float q = xf[m * (size_t)Nn + n] + pe[(size_t)b * Nn + n] * gA[n] + v * gB[n];
        Cbf[m * Nn + n] = (__bf16)q;
      } else if (mode == 2) {
        int b = (int)(m / NN), i2 = (int)(m % NN);
        int h = n / DH, d = n % DH;
        Cbf[(((size_t)(b * HH + h)) * NN + i2) * DH + d] = (__bf16)v;
      } else {
        Cf[m * (size_t)Nn + n] = v + bias[n];
      }
    }
}

// ---------------------------------------------------------------------------
// Fused attention per (b,h): 128 query rows/block, 8 waves, K/V in LDS via
// async loads. QK^T (WMMA) -> in-register softmax (shfl_xor over 16-lane
// halves) -> P in LDS -> P@V (WMMA) -> O[b][i][h*64+d] bf16.
// ---------------------------------------------------------------------------
__global__ __launch_bounds__(256)
void attention(const __bf16* __restrict__ Q, const __bf16* __restrict__ Kt,
               const __bf16* __restrict__ Vst, __bf16* __restrict__ O) {
  __shared__ __align__(16) __bf16 Ks[JP * DH];       // [j][d]
  __shared__ __align__(16) __bf16 Vs[DH * JP];       // [d][j]
  __shared__ __align__(16) __bf16 Ps[8 * 16 * JP];   // per-wave [16][jp]
  int bh = blockIdx.x;
  int m0 = blockIdx.y * 128;
  int t = threadIdx.x, w = t >> 5, lane = t & 31;
  int idx = lane & 15, half = lane >> 4;

  {  // async stage K and V^T for this (b,h): 768 + 768 chunks of 16B
    const __bf16* Kg = Kt + (size_t)bh * JP * DH;
    const __bf16* Vg = Vst + (size_t)bh * DH * JP;
    for (int c = t; c < (JP * DH) / 8; c += 256) {
      uint32_t go = (uint32_t)(c * 16);
      async_ld_b128(Ks + c * 8, Kg, go);
      async_ld_b128(Vs + c * 8, Vg, go);
    }
    wait_async0();
  }
  __syncthreads();

  const __bf16* Qg = Q + (size_t)bh * NN * DH;
  int r0 = m0 + (w << 4);

  v8f sim[JT];
#pragma unroll
  for (int jt = 0; jt < JT; ++jt)
#pragma unroll
    for (int r = 0; r < 8; ++r) sim[jt][r] = 0.f;

#pragma unroll
  for (int kb = 0; kb < DH; kb += 32) {
    v16bf a = frag_ld(Qg, DH, r0, kb);
#pragma unroll
    for (int jt = 0; jt < JT; ++jt) {
      v16bf bb = frag_ld(Ks, DH, jt * 16, kb);
      sim[jt] = wmma_bf16(a, bb, sim[jt]);
    }
  }

  // softmax over j per row; row M = r + 8*half lives across one 16-lane half
#pragma unroll
  for (int r = 0; r < 8; ++r) {
    float mx = -1e30f;
#pragma unroll
    for (int jt = 0; jt < JT; ++jt) {
      int j = jt * 16 + idx;
      float v = sim[jt][r] * ATTN_SCALE;
      v = (j < JJ) ? v : -1e30f;
      sim[jt][r] = v;
      mx = fmaxf(mx, v);
    }
#pragma unroll
    for (int o = 8; o; o >>= 1) mx = fmaxf(mx, __shfl_xor(mx, o, 32));
    float sum = 0.f;
#pragma unroll
    for (int jt = 0; jt < JT; ++jt) {
      float e = __expf(sim[jt][r] - mx);
      sim[jt][r] = e;
      sum += e;
    }
#pragma unroll
    for (int o = 8; o; o >>= 1) sum += __shfl_xor(sum, o, 32);
    float inv = 1.0f / sum;
    int m = r + 8 * half;
#pragma unroll
    for (int jt = 0; jt < JT; ++jt) {
      int j = jt * 16 + idx;
      Ps[((w << 4) + m) * JP + j] = (__bf16)(sim[jt][r] * inv);
    }
  }
  __syncthreads();

  v8f o4[4];
#pragma unroll
  for (int s = 0; s < 4; ++s)
#pragma unroll
    for (int r = 0; r < 8; ++r) o4[s][r] = 0.f;

#pragma unroll
  for (int kb = 0; kb < JP; kb += 32) {
    v16bf a = frag_ld(Ps + (size_t)(w << 4) * JP, JP, 0, kb);
#pragma unroll
    for (int s = 0; s < 4; ++s) {
      v16bf bb = frag_ld(Vs, JP, s * 16, kb);
      o4[s] = wmma_bf16(a, bb, o4[s]);
    }
  }

  int b = bh / HH, h = bh % HH;
#pragma unroll
  for (int s = 0; s < 4; ++s)
#pragma unroll
    for (int r = 0; r < 8; ++r) {
      int i = r0 + r + 8 * half;
      int d = s * 16 + idx;
      O[((size_t)b * NN + i) * QD + h * DH + d] = (__bf16)o4[s][r];
    }
}

// ---------------------------------------------------------------------------
extern "C" void kernel_launch(void* const* d_in, const int* in_sizes, int n_in,
                              void* d_out, int out_size, void* d_ws, size_t ws_size,
                              hipStream_t stream) {
  const float* x       = (const float*)d_in[0];
  const float* ctx     = (const float*)d_in[1];
  const int*   ctype   = (const int*)d_in[2];
  const float* prog    = (const float*)d_in[3];
  const float* tt_emb  = (const float*)d_in[4];
  const float* tt_m1_w = (const float*)d_in[5];
  const float* tt_m1_b = (const float*)d_in[6];
  const float* tt_m2_w = (const float*)d_in[7];
  const float* tt_m2_b = (const float*)d_in[8];
  const float* tt_gA   = (const float*)d_in[9];
  const float* tt_gB   = (const float*)d_in[10];
  const float* pe_p1_w = (const float*)d_in[11];
  const float* pe_p1_b = (const float*)d_in[12];
  const float* pe_p2_w = (const float*)d_in[13];
  const float* pe_p2_b = (const float*)d_in[14];
  const float* pe_m1_w = (const float*)d_in[15];
  const float* pe_m1_b = (const float*)d_in[16];
  const float* pe_m2_w = (const float*)d_in[17];
  const float* pe_m2_b = (const float*)d_in[18];
  const float* pe_gA   = (const float*)d_in[19];
  const float* pe_gB   = (const float*)d_in[20];
  const float* wq      = (const float*)d_in[21];
  const float* wk      = (const float*)d_in[22];
  const float* wv      = (const float*)d_in[23];
  const float* wo      = (const float*)d_in[24];
  const float* bo      = (const float*)d_in[25];
  float* out = (float*)d_out;

  char* wsp = (char*)d_ws;
  size_t off = 0;
  auto alloc = [&](size_t bytes) -> void* {
    void* p = wsp + off;
    off += (bytes + 255) & ~(size_t)255;
    return p;
  };

  const size_t M = (size_t)BB * NN;                       // 65536 query rows
  __bf16* ctx_bf = (__bf16*)alloc((size_t)BB * JJ * QD * 2);
  __bf16* kin_bf = (__bf16*)alloc((size_t)BB * JJ * QD * 2);
  float*  pe     = (float*) alloc((size_t)BB * QD * 4);
  __bf16* wkt    = (__bf16*)alloc((size_t)QD * QD * 2);
  __bf16* wvt    = (__bf16*)alloc((size_t)QD * QD * 2);
  __bf16* wqt    = (__bf16*)alloc((size_t)QD * QD * 2);
  __bf16* wot    = (__bf16*)alloc((size_t)QD * QD * 2);
  __bf16* m1t    = (__bf16*)alloc((size_t)QD * (QD / 2) * 2);
  __bf16* m2t    = (__bf16*)alloc((size_t)(QD / 2) * QD * 2);
  __bf16* Ktb    = (__bf16*)alloc((size_t)BB * HH * JP * DH * 2);
  __bf16* Vst    = (__bf16*)alloc((size_t)BB * HH * DH * JP * 2);
  __bf16* buf1   = (__bf16*)alloc(M * QD * 2);            // xpe -> Q
  __bf16* buf2   = (__bf16*)alloc(M * (QD / 2) * 2);      // h1
  __bf16* buf3   = (__bf16*)alloc(M * QD * 2);            // qin -> O

  // 1) weight convert + transpose to bf16 [N][K]
  int blk = 256;
  cvt_transpose<<<(QD * QD + blk - 1) / blk, blk, 0, stream>>>(wk, wkt, QD, QD);
  cvt_transpose<<<(QD * QD + blk - 1) / blk, blk, 0, stream>>>(wv, wvt, QD, QD);
  cvt_transpose<<<(QD * QD + blk - 1) / blk, blk, 0, stream>>>(wq, wqt, QD, QD);
  cvt_transpose<<<(QD * QD + blk - 1) / blk, blk, 0, stream>>>(wo, wot, QD, QD);
  cvt_transpose<<<(QD * (QD / 2) + blk - 1) / blk, blk, 0, stream>>>(pe_m1_w, m1t, QD, QD / 2);
  cvt_transpose<<<((QD / 2) * QD + blk - 1) / blk, blk, 0, stream>>>(pe_m2_w, m2t, QD / 2, QD);

  // 2) context-side MLP + gating; progress embedding
  prep_context<<<BB * JJ, 256, 0, stream>>>(ctx, ctype, tt_emb, tt_m1_w, tt_m1_b,
                                            tt_m2_w, tt_m2_b, tt_gA, tt_gB,
                                            ctx_bf, kin_bf);
  prep_pe<<<BB, 256, 0, stream>>>(prog, pe_p1_w, pe_p1_b, pe_p2_w, pe_p2_b, pe);

  // 3) K / V projections into attention layouts
  kv_gemm<<<dim3(BB * JT, HH), 32, 0, stream>>>(kin_bf, ctx_bf, wkt, wvt, Ktb, Vst);

  // 4) xpe = bf16(x + pe[b])
  make_xpe<<<(unsigned)((M * QD) / 256), 256, 0, stream>>>(x, pe, buf1);

  // 5) h1 = gelu(xpe @ pe_m1_w + b1)            (mode 0)
  gemm_bf16<<<dim3(M / 128, (QD / 2) / 128), 256, 0, stream>>>(
      buf1, m1t, QD, QD / 2, 0, pe_m1_b, nullptr, nullptr, nullptr, nullptr,
      buf2, nullptr);

  // 6) qin = x + pe*gA + (h1 @ pe_m2_w + b2)*gB (mode 1)
  gemm_bf16<<<dim3(M / 128, QD / 128), 256, 0, stream>>>(
      buf2, m2t, QD / 2, QD, 1, pe_m2_b, x, pe, pe_gA, pe_gB, buf3, nullptr);

  // 7) Q = qin @ wq, reshaped [b][h][i][d]      (mode 2)
  gemm_bf16<<<dim3(M / 128, QD / 128), 256, 0, stream>>>(
      buf3, wqt, QD, QD, 2, nullptr, nullptr, nullptr, nullptr, nullptr,
      buf1, nullptr);

  // 8) attention -> O[b][i][h*64+d] bf16
  attention<<<dim3(BB * HH, NN / 128), 256, 0, stream>>>(buf1, Ktb, Vst, buf3);

  // 9) out = O @ wo + bo  (f32)                 (mode 3)
  gemm_bf16<<<dim3(M / 128, QD / 128), 256, 0, stream>>>(
      buf3, wot, QD, QD, 3, bo, nullptr, nullptr, nullptr, nullptr,
      nullptr, out);
}
